// pdf_79980880986280
// MI455X (gfx1250) — matrix operand order repaired and checked by reading, not compiled, not run on hardware
//
#include <hip/hip_runtime.h>

// ---------------------------------------------------------------------------
// PDF / RDF soft-histogram for BaTiO3, CDNA5 (gfx1250) WMMA formulation.
//   counts[t,b] = sum_p sel[t,p] * exp(-0.5*((dist_p - c_b)/sigma)^2)
// computed as tiled 16x16x32 f16 WMMAs: A = one-hot type selector (exact in
// f16), B = Gaussian kernel tile, C = f32 accumulators.
// Gaussian evaluated as exp2(-(s*d - s*c)^2) with s = sqrt(0.5*log2(e))/sigma,
// distances pre-scaled by s at LDS staging, so the hot loop per value is just
// v_sub + v_mul (neg folded) feeding v_exp_f32.
// ---------------------------------------------------------------------------

typedef __attribute__((ext_vector_type(16))) _Float16 v16h;
typedef __attribute__((ext_vector_type(8)))  float    v8f;

#define NATOMS   1024
#define NBINS    100
#define NBINS_P  112          // padded to 7 WMMA tiles of 16
#define NTILES   7
#define NTYPES   6
#define R0C      0.5f
#define WIDTHC   0.07f        // (7.5-0.5)/100
#define CUTOFFC  8.0f
#define BLK      64           // atom block edge -> 16 blocks, 136 block-pairs
#define NBLOCKS  16
#define NBPAIRS  136          // 16*17/2
#define CNT_STRIDE (NTYPES * NBINS_P)   // 672 floats per frame

// invSigma = sqrt(2*pi)/width ; k2 = -0.5*invSigma^2*log2(e) ; s = sqrt(-k2)
#define INV_SIGMA_D (2.50662827463100050242 / 0.07)
#define SQK_D (INV_SIGMA_D * 0.84932180028801904272)  // sqrt(0.5*log2(e))
#define SQKC ((float)SQK_D)
#define PADCTR ((float)(SQK_D * 4.0e9))   // scaled fake center for pad bins

__global__ void zero_ws_kernel(float* ws, int n) {
    int i = blockIdx.x * blockDim.x + threadIdx.x;
    if (i < n) ws[i] = 0.0f;
}

__global__ __launch_bounds__(256)
void pdf_pair_wmma_kernel(const float* __restrict__ xyz,
                          const int*   __restrict__ numbers,
                          const float* __restrict__ cell,
                          float*       __restrict__ wsCounts) {
    __shared__ float         sPos[2][BLK][3];
    __shared__ int           sTyp[2][BLK];
    __shared__ float         sDist[8][32];   // holds s * dist
    __shared__ unsigned char sCode[8][32];

    const int tid  = threadIdx.x;
    const int wave = tid >> 5;
    const int lane = tid & 31;
    const int row  = lane & 15;
    const bool hi  = lane >= 16;

    // decode (frame, bi, bj) from blockIdx
    const int frame = blockIdx.x / NBPAIRS;
    int u = blockIdx.x % NBPAIRS;
    int bi = 0;
    while (u >= (NBLOCKS - bi)) { u -= (NBLOCKS - bi); ++bi; }
    const int bj = bi + u;

    const float Lc   = cell[0];
    const float invL = 1.0f / Lc;

    // stage the two 64-atom blocks (positions + type codes 0:O 1:Ti 2:Ba)
    if (tid < 2 * BLK) {
        const int side = tid / BLK;           // 0 = i-block, 1 = j-block
        const int a    = tid % BLK;
        const int atom = (side ? bj : bi) * BLK + a;
        const int base = frame * NATOMS * 3 + atom * 3;
        sPos[side][a][0] = xyz[base + 0];
        sPos[side][a][1] = xyz[base + 1];
        sPos[side][a][2] = xyz[base + 2];
        const int z = numbers[atom];
        sTyp[side][a] = (z == 8) ? 0 : ((z == 22) ? 1 : 2);
    }
    __syncthreads();

    // per-lane *scaled* bin centers for the 7 tiles (column = row);
    // padded bins >= 100 get a huge fake center so exp2 underflows to 0.
    float ctrs[NTILES];
#pragma unroll
    for (int tile = 0; tile < NTILES; ++tile) {
        const int bin = tile * 16 + row;
        ctrs[tile] = (bin < NBINS)
                         ? (SQKC * (R0C + WIDTHC * ((float)bin + 0.5f)))
                         : PADCTR;
    }

    v8f acc[NTILES];
    const v8f vzero = {0.f, 0.f, 0.f, 0.f, 0.f, 0.f, 0.f, 0.f};
#pragma unroll
    for (int t = 0; t < NTILES; ++t) acc[t] = vzero;

    // each wave: 16 chunks of 32 pairs within the 64x64 tile
    for (int c = 0; c < 16; ++c) {
        const int q  = wave * 512 + c * 32 + lane;   // 0..4095
        const int il = q >> 6;
        const int jl = q & 63;

        float dx = sPos[0][il][0] - sPos[1][jl][0];
        float dy = sPos[0][il][1] - sPos[1][jl][1];
        float dz = sPos[0][il][2] - sPos[1][jl][2];
        dx -= Lc * rintf(dx * invL);
        dy -= Lc * rintf(dy * invL);
        dz -= Lc * rintf(dz * invL);
        const float dist = sqrtf(dx * dx + dy * dy + dz * dz);

        const int gi = bi * BLK + il;
        const int gj = bj * BLK + jl;
        const bool valid = (gi < gj) && (dist < CUTOFFC);

        const int ti = sTyp[0][il], tj = sTyp[1][jl];
        const int a0 = ti < tj ? ti : tj;
        const int b0 = ti < tj ? tj : ti;
        const int ptype = a0 * (7 - a0) / 2 + (b0 - a0);   // 0..5

        sCode[wave][lane] = valid ? (unsigned char)ptype : (unsigned char)255;
        sDist[wave][lane] = SQKC * dist;   // pre-scaled distance
        __builtin_amdgcn_wave_barrier();   // same-wave LDS ops are in-order

        // ---- A fragment: 16x32 f16 one-hot selector -----------------------
        // lanes 0-15 hold K in {0..7, 16..23}; lanes 16-31 K in {8..15, 24..31}
        const int kbA = hi ? 8 : 0;
        v16h afrag;
#pragma unroll
        for (int h = 0; h < 8; ++h) {
            afrag[h]     = (sCode[wave][kbA + h]      == (unsigned char)row)
                               ? (_Float16)1.0f : (_Float16)0.0f;
            afrag[h + 8] = (sCode[wave][kbA + 16 + h] == (unsigned char)row)
                               ? (_Float16)1.0f : (_Float16)0.0f;
        }

        // gather the 16 scaled distances this lane needs for its B half
        const int kbB = hi ? 8 : 0;
        float dk[16];
#pragma unroll
        for (int h = 0; h < 16; ++h) dk[h] = sDist[wave][kbB + (h & 7) + ((h >> 3) << 4)];

        // ---- B fragments + WMMA accumulate over 7 bin tiles ---------------
#pragma unroll
        for (int tile = 0; tile < NTILES; ++tile) {
            const float ctr = ctrs[tile];
            v16h bfrag;
#pragma unroll
            for (int h = 0; h < 16; ++h) {
                const float t = dk[h] - ctr;
                const float g = __builtin_amdgcn_exp2f(-(t * t));
                bfrag[h] = (_Float16)g;
            }
            acc[tile] = __builtin_amdgcn_wmma_f32_16x16x32_f16(
                false, afrag, false, bfrag, (short)0, acc[tile], false, false);
        }
    }

    // ---- merge: C layout -> lanes 0-15 hold M=0..7 in VGPR 0..7 ----------
    if (lane < 16) {
        float* dst = wsCounts + frame * CNT_STRIDE;
#pragma unroll
        for (int r = 0; r < NTYPES; ++r) {
#pragma unroll
            for (int tile = 0; tile < NTILES; ++tile) {
                atomicAdd(&dst[r * NBINS_P + tile * 16 + lane], acc[tile][r]);
            }
        }
    }
}

__global__ void pdf_finalize_kernel(const float* __restrict__ wsCounts,
                                    const float* __restrict__ bins,
                                    float*       __restrict__ out) {
    __shared__ float S[2][NTYPES];
    const int tid = threadIdx.x;   // 128 threads

    if (tid < 2 * NTYPES) {
        const int f = tid / NTYPES, t = tid % NTYPES;
        float s = 0.0f;
        for (int b = 0; b < NBINS; ++b)
            s += wsCounts[f * CNT_STRIDE + t * NBINS_P + b];
        S[f][t] = s;
    }
    __syncthreads();

    if (tid < NBINS) {
        const float PI = 3.14159265358979323846f;
        const float r0 = R0C + WIDTHC * (float)tid;
        const float r1 = r0 + WIDTHC;
        const float vol = (4.0f * PI / 3.0f) * (r1 * r1 * r1 - r0 * r0 * r0);
        const float V   = (4.0f / 3.0f) * PI * 7.5f * 7.5f * 7.5f;
        const float vinv = V / vol;
        const float coeff[NTYPES] = {23.04f, 42.24f, 107.52f,
                                     19.36f, 98.56f, 125.44f};
        float acc = 0.0f;
        for (int f = 0; f < 2; ++f)
            for (int t = 0; t < NTYPES; ++t) {
                const float cn =
                    wsCounts[f * CNT_STRIDE + t * NBINS_P + tid] / S[f][t];
                acc += coeff[t] * (cn * vinv - 1.0f);
            }
        out[tid] = acc * 0.5f / 416.16f;   // mean over 2 frames, then /416.16
    }
    if (tid < NBINS + 1) out[NBINS + tid] = bins[tid];   // bins passthrough
}

extern "C" void kernel_launch(void* const* d_in, const int* in_sizes, int n_in,
                              void* d_out, int out_size, void* d_ws, size_t ws_size,
                              hipStream_t stream) {
    const float* xyz     = (const float*)d_in[0];  // [2][1024][3]
    const int*   numbers = (const int*)d_in[1];    // [1024]
    const float* bins    = (const float*)d_in[2];  // [101]
    const float* cell    = (const float*)d_in[3];  // [3]
    float*       ws      = (float*)d_ws;           // [2][6][112] counts
    float*       out     = (float*)d_out;          // [100] pdf + [101] bins

    const int nCnt = 2 * CNT_STRIDE;               // 1344 floats
    zero_ws_kernel<<<(nCnt + 255) / 256, 256, 0, stream>>>(ws, nCnt);
    pdf_pair_wmma_kernel<<<2 * NBPAIRS, 256, 0, stream>>>(xyz, numbers, cell, ws);
    pdf_finalize_kernel<<<1, 128, 0, stream>>>(ws, bins, out);
    (void)in_sizes; (void)n_in; (void)out_size; (void)ws_size;
}
